// DiffusionModel_88948772700540
// MI455X (gfx1250) — matrix operand order repaired and checked by reading, not compiled, not run on hardware
//
#include <hip/hip_runtime.h>

#ifndef __has_builtin
#define __has_builtin(x) 0
#endif

// ---------------------------------------------------------------------------
// Types for CDNA5 WMMA
// ---------------------------------------------------------------------------
typedef __attribute__((ext_vector_type(16))) __bf16 v16bf;
typedef __attribute__((ext_vector_type(8)))  float  v8f;
typedef __attribute__((ext_vector_type(4)))  float  f32x4;

// Model constants
#define MB  16      // batch
#define MN  64      // nodes
#define NHD 256
#define EHD 128
#define CHD 256
#define NHEADS 8

#if __has_builtin(__builtin_amdgcn_cvt_pk_bf16_f32)
typedef __attribute__((ext_vector_type(2))) __bf16 v2bf;
static __device__ __forceinline__ unsigned pk_bf16(float a, float b)
{
    union { v2bf v; unsigned u; } x;
    x.v = __builtin_amdgcn_cvt_pk_bf16_f32(a, b);
    return x.u;
}
#else
static __device__ __forceinline__ unsigned pk_bf16(float a, float b)
{
    union { __bf16 h[2]; unsigned u; } x;
    x.h[0] = (__bf16)a; x.h[1] = (__bf16)b;
    return x.u;
}
#endif

// ---------------------------------------------------------------------------
// Generic GEMM: Out[M,N] = act(A[M,K] @ W[K,N] + bias) (+ Res)
// bf16 WMMA 16x16x32, f32 accumulate. Block = 8 waves, tile 128(M) x 64(N).
// FULL=true: M%128==0, N%64==0, K%32==0 -> no guards, vectorized staging.
// ---------------------------------------------------------------------------
#define BM 128
#define BN 64
#define BK 32
#define ASTR 36   // LDS row stride (bf16): 72B rows -> 8B aligned, odd 8B-groups
#define BSTR 36

template <bool FULL>
__global__ __launch_bounds__(256)
void gemm_bf16_wmma_kernel(const float* __restrict__ A,
                           const float* __restrict__ W,
                           const float* __restrict__ bias,
                           const float* __restrict__ Res,
                           float* __restrict__ C,
                           int M, int N, int K, int act)
{
    __shared__ __align__(16) __bf16 As[BM * ASTR];
    __shared__ __align__(16) __bf16 Bs[BN * BSTR];

    const int t    = threadIdx.x;
    const int wave = t >> 5;
    const int lane = t & 31;
    const int half = lane >> 4;
    const int l16  = lane & 15;

    const int m0 = blockIdx.y * BM;
    const int n0 = blockIdx.x * BN;

    v8f acc0 = {}; v8f acc1 = {}; v8f acc2 = {}; v8f acc3 = {};

    for (int k0 = 0; k0 < K; k0 += BK) {
        if (FULL) {
            // ---- issue ALL tile loads first so they overlap (one wait)
            f32x4 av[4];
            const int ar  = t >> 3;          // base row within 32-row group
            const int ac4 = (t & 7) * 4;
#pragma unroll
            for (int i = 0; i < 4; ++i)
                av[i] = *(const f32x4*)(A + (long)(m0 + i * 32 + ar) * K + (k0 + ac4));

            f32x4 wv4[2];
            const int wk0 = (t + 0 * 256) >> 4, wn0 = ((t + 0 * 256) & 15) * 4;
            const int wk1 = (t + 1 * 256) >> 4, wn1 = ((t + 1 * 256) & 15) * 4;
            wv4[0] = *(const f32x4*)(W + (long)(k0 + wk0) * N + (n0 + wn0));
            wv4[1] = *(const f32x4*)(W + (long)(k0 + wk1) * N + (n0 + wn1));

            // ---- convert + store to LDS (A: packed bf16 pairs; W: transposed)
#pragma unroll
            for (int i = 0; i < 4; ++i) {
                uint2 pk;
                pk.x = pk_bf16(av[i].x, av[i].y);
                pk.y = pk_bf16(av[i].z, av[i].w);
                *(uint2*)(&As[(i * 32 + ar) * ASTR + ac4]) = pk;
            }
            Bs[(wn0 + 0) * BSTR + wk0] = (__bf16)wv4[0].x;
            Bs[(wn0 + 1) * BSTR + wk0] = (__bf16)wv4[0].y;
            Bs[(wn0 + 2) * BSTR + wk0] = (__bf16)wv4[0].z;
            Bs[(wn0 + 3) * BSTR + wk0] = (__bf16)wv4[0].w;
            Bs[(wn1 + 0) * BSTR + wk1] = (__bf16)wv4[1].x;
            Bs[(wn1 + 1) * BSTR + wk1] = (__bf16)wv4[1].y;
            Bs[(wn1 + 2) * BSTR + wk1] = (__bf16)wv4[1].z;
            Bs[(wn1 + 3) * BSTR + wk1] = (__bf16)wv4[1].w;

            if (k0 + BK < K) {   // uniform branch: prefetch next tiles
                __builtin_prefetch(A + (long)(m0 + ar) * K + (k0 + BK + ac4), 0, 3);
                __builtin_prefetch(W + (long)(k0 + BK + wk0) * N + (n0 + wn0), 0, 3);
            }
        } else {
            // ---- guarded scalar staging (small / ragged GEMMs only)
#pragma unroll
            for (int i = 0; i < (BM * BK) / 256; ++i) {
                int idx = t + i * 256;
                int r = idx >> 5;
                int c = idx & 31;
                int gm = m0 + r, gk = k0 + c;
                float v = 0.f;
                if (gm < M && gk < K) v = A[(long)gm * K + gk];
                As[r * ASTR + c] = (__bf16)v;
            }
#pragma unroll
            for (int i = 0; i < (BK * BN) / 256; ++i) {
                int idx = t + i * 256;
                int kk = idx >> 6;
                int nn = idx & 63;
                int gk = k0 + kk, gn = n0 + nn;
                float v = 0.f;
                if (gk < K && gn < N) v = W[(long)gk * N + gn];
                Bs[nn * BSTR + kk] = (__bf16)v;
            }
        }
        __syncthreads();

        // ---- A fragment (16x32 bf16): row = wave*16 + l16, K pairs per ISA layout
        union { v16bf v; unsigned u[8]; } af;
        {
            const unsigned* ap = (const unsigned*)(As + (wave * 16 + l16) * ASTR);
#pragma unroll
            for (int p = 0; p < 8; ++p) {
                int kk = 2 * p + 8 * half + ((p >= 4) ? 8 : 0);
                af.u[p] = ap[kk >> 1];
            }
        }
        // ---- B fragments (32x16 bf16) + WMMA
#pragma unroll
        for (int nt = 0; nt < 4; ++nt) {
            union { v16bf v; unsigned u[8]; } bf;
            const unsigned* bp = (const unsigned*)(Bs + (nt * 16 + l16) * BSTR);
#pragma unroll
            for (int p = 0; p < 8; ++p) {
                int kk = 2 * p + 8 * half + ((p >= 4) ? 8 : 0);
                bf.u[p] = bp[kk >> 1];
            }
            v8f* accp = (nt == 0) ? &acc0 : (nt == 1) ? &acc1 : (nt == 2) ? &acc2 : &acc3;
            *accp = __builtin_amdgcn_wmma_f32_16x16x32_bf16(
                        false, af.v, false, bf.v, (short)0, *accp, false, false);
        }
        __syncthreads();
    }

    // ---- epilogue: bias + activation + residual
#pragma unroll
    for (int nt = 0; nt < 4; ++nt) {
        const v8f* accp = (nt == 0) ? &acc0 : (nt == 1) ? &acc1 : (nt == 2) ? &acc2 : &acc3;
        int gn = n0 + nt * 16 + l16;
#pragma unroll
        for (int r = 0; r < 8; ++r) {
            int gm = m0 + wave * 16 + r + 8 * half;
            if (FULL || (gm < M && gn < N)) {
                float v = (*accp)[r];
                if (bias) v += bias[gn];
                if (act)  v = v > 0.f ? v : 0.1f * v;
                if (Res)  v += Res[(long)gm * N + gn];
                C[(long)gm * N + gn] = v;
            }
        }
    }
}

// ---------------------------------------------------------------------------
// LayerNorm (no affine) + optional AdaLN modulation (per-batch) + LeakyReLU
// Row staged to LDS via CDNA5 async global->LDS copy (ASYNCcnt) when the
// builtin is available; plain LDS store fallback otherwise.
// NOTE: builtin parameter types are addrspace(1)/(3) int* per clang-22.
// ---------------------------------------------------------------------------
__global__ __launch_bounds__(128)
void ln_mod_kernel(const float* __restrict__ x, float* __restrict__ out,
                   const float* __restrict__ mvec, const float* __restrict__ avec,
                   int C, int rows_per_batch, int mod_stride)
{
    __shared__ float xs[256];
    long row = blockIdx.x;
    int  b   = (int)(row / rows_per_batch);
    const float* xr = x + row * C;
    float*       orow = out + row * C;
    int t = threadIdx.x;

#if __has_builtin(__builtin_amdgcn_global_load_async_to_lds_b32)
    {
        typedef __attribute__((address_space(1))) int gint_t;
        typedef __attribute__((address_space(3))) int lint_t;
        __builtin_amdgcn_global_load_async_to_lds_b32(
            (gint_t*)(unsigned long long)(xr + t),
            (lint_t*)(unsigned)(unsigned long long)&xs[t], 0, 0);
        if (C > 128)
            __builtin_amdgcn_global_load_async_to_lds_b32(
                (gint_t*)(unsigned long long)(xr + t + 128),
                (lint_t*)(unsigned)(unsigned long long)&xs[t + 128], 0, 0);
#if __has_builtin(__builtin_amdgcn_s_wait_asynccnt)
        __builtin_amdgcn_s_wait_asynccnt(0);
#else
        asm volatile("s_wait_asynccnt 0x0" ::: "memory");
#endif
    }
#else
    xs[t] = xr[t];
    if (C > 128) xs[t + 128] = xr[t + 128];
#endif
    __syncthreads();

    float s = 0.f, s2 = 0.f;
    for (int j = t; j < C; j += 128) { float v = xs[j]; s += v; s2 += v * v; }
    for (int off = 16; off; off >>= 1) { s += __shfl_xor(s, off); s2 += __shfl_xor(s2, off); }
    __shared__ float red[8];
    int wave = t >> 5, lane = t & 31;
    if (lane == 0) { red[wave] = s; red[4 + wave] = s2; }
    __syncthreads();
    s  = red[0] + red[1] + red[2] + red[3];
    s2 = red[4] + red[5] + red[6] + red[7];
    float mean = s / C;
    float var  = s2 / C - mean * mean;
    float rstd = rsqrtf(var + 1e-5f);
    for (int j = t; j < C; j += 128) {
        float v = (xs[j] - mean) * rstd;
        if (mvec) v = v * (1.f + mvec[(long)b * mod_stride + j]) + avec[(long)b * mod_stride + j];
        v = v > 0.f ? v : 0.1f * v;
        orow[j] = v;
    }
}

// ---------------------------------------------------------------------------
// Attention elementwise: one block per (b,i,j); 256 thr = (h=t>>5, d=t&31)
// ne = q*k*scale + q*k + shift ; S[b,h,j,i] = sum_d ne / sqrt(32)
// Each wave == one head -> full-wave shfl reduction gives head sum.
// ---------------------------------------------------------------------------
__global__ __launch_bounds__(256)
void attn_kernel(const float* __restrict__ qkv, const float* __restrict__ ss,
                 float* __restrict__ ne_leaky, float* __restrict__ S)
{
    int b = blockIdx.y;
    int ij = blockIdx.x;
    int i = ij / MN, j = ij % MN;
    int t = threadIdx.x;
    int h = t >> 5;

    float q  = qkv[((long)(b * MN + i) * (3 * NHD)) + t];
    float k  = qkv[((long)(b * MN + j) * (3 * NHD)) + NHD + t];
    long  eb = (long)(b * MN + i) * MN + j;
    float sh = ss[eb * (2 * NHD) + t];
    float sc = ss[eb * (2 * NHD) + NHD + t];
    float at = q * k;
    float ne = at * sc + at + sh;

    float sum = ne;
    for (int off = 16; off; off >>= 1) sum += __shfl_xor(sum, off);
    if ((t & 31) == 0)
        S[(((long)(b * NHEADS + h) * MN + j) * MN) + i] = sum * 0.17677669529663687f; // 1/sqrt(32)
    ne_leaky[eb * NHD + t] = ne > 0.f ? ne : 0.1f * ne;
}

// ---------------------------------------------------------------------------
// Softmax over i (last dim of S[b,h,j,i]); one 64-thread block per row
// ---------------------------------------------------------------------------
__global__ __launch_bounds__(64)
void softmax_kernel(const float* __restrict__ S, float* __restrict__ Aout)
{
    long row = blockIdx.x;
    int t = threadIdx.x;
    __shared__ float sh[64];
    float v = S[row * MN + t];
    sh[t] = v; __syncthreads();
    for (int off = 32; off; off >>= 1) { if (t < off) sh[t] = fmaxf(sh[t], sh[t + off]); __syncthreads(); }
    float mx = sh[0]; __syncthreads();
    float e = __expf(v - mx);
    sh[t] = e; __syncthreads();
    for (int off = 32; off; off >>= 1) { if (t < off) sh[t] += sh[t + off]; __syncthreads(); }
    float sum = sh[0];
    Aout[row * MN + t] = e / sum;
}

// ---------------------------------------------------------------------------
// wv[b,j,h*32+d] = sum_i A[b,h,j,i] * v[b,i, 512 + h*32+d]
// A is wave-uniform (one head per wave) -> scalar broadcast per iter
// ---------------------------------------------------------------------------
__global__ __launch_bounds__(256)
void wv_kernel(const float* __restrict__ Asm, const float* __restrict__ qkv,
               float* __restrict__ wv)
{
    int b = blockIdx.y, j = blockIdx.x;
    int t = threadIdx.x, h = t >> 5;
    const float* arow = Asm + ((long)(b * NHEADS + h) * MN + j) * MN;
    float acc = 0.f;
    for (int i = 0; i < MN; ++i)
        acc = fmaf(arow[i], qkv[((long)(b * MN + i) * (3 * NHD)) + 2 * NHD + t], acc);
    wv[((long)(b * MN + j) * NHD) + t] = acc;
}

// ---------------------------------------------------------------------------
// Small helpers
// ---------------------------------------------------------------------------
__global__ void posadd_kernel(float* __restrict__ nodes, const float* __restrict__ pos)
{
    long idx = (long)blockIdx.x * blockDim.x + threadIdx.x;   // over B*N*NHD
    int c = (int)(idx % NHD);
    int n = (int)((idx / NHD) % MN);
    nodes[idx] += pos[n * NHD + c];
}

__global__ void tgather_kernel(const int* __restrict__ ts, const float* __restrict__ temb,
                               float* __restrict__ te)
{
    int b = blockIdx.x, c = threadIdx.x;
    te[b * CHD + c] = temb[(long)ts[b] * CHD + c];
}

__global__ void assemble_nodes_kernel(const float* __restrict__ types,
                                      const float* __restrict__ prms,
                                      float* __restrict__ out)
{
    long row = blockIdx.x;
    int c = threadIdx.x;
    if (c >= 21) return;
    float v;
    if (c == 0)      v = 0.f;
    else if (c < 7)  v = types[row * 6 + (c - 1)];
    else             v = prms[row * 14 + (c - 7)];
    out[row * 21 + c] = v;
}

// ---------------------------------------------------------------------------
// Host orchestration
// ---------------------------------------------------------------------------
static inline void launch_gemm(const float* A, const float* W, const float* bias,
                               const float* Res, float* Out,
                               int M, int N, int K, int act, hipStream_t s)
{
    dim3 g((N + BN - 1) / BN, (M + BM - 1) / BM);
    if ((M % BM) == 0 && (N % BN) == 0 && (K % BK) == 0)
        gemm_bf16_wmma_kernel<true ><<<g, dim3(256), 0, s>>>(A, W, bias, Res, Out, M, N, K, act);
    else
        gemm_bf16_wmma_kernel<false><<<g, dim3(256), 0, s>>>(A, W, bias, Res, Out, M, N, K, act);
}

extern "C" void kernel_launch(void* const* d_in, const int* in_sizes, int n_in,
                              void* d_out, int out_size, void* d_ws, size_t ws_size,
                              hipStream_t stream)
{
    (void)in_sizes; (void)n_in; (void)out_size; (void)ws_size;

    const float* nodes_in = (const float*)d_in[0];   // [16,64,21]
    const float* edges_in = (const float*)d_in[1];   // [16,64,64,5]
    const int*   timestep = (const int*)d_in[2];     // [16]

    auto F = [&](int i) { return (const float*)d_in[i]; };
    int pi = 3;

    // JAX pytree leaf order: sorted dict keys; 'b' < 'w', 'l0' < 'l1'.
    struct Layer {
        const float *eo_b,*eo_w,*lc_b,*lc_w,*me0_b,*me0_w,*me1_b,*me1_w,
                    *mn0_b,*mn0_w,*mn1_b,*mn1_w,*no_b,*no_w,*qkv_b,*qkv_w,*ss_b,*ss_w;
    } L[4];
    for (int l = 0; l < 4; ++l) {
        L[l].eo_b  = F(pi++); L[l].eo_w  = F(pi++);   // edges_out
        L[l].lc_b  = F(pi++); L[l].lc_w  = F(pi++);   // lin_cond
        L[l].me0_b = F(pi++); L[l].me0_w = F(pi++);   // mlp_edges.l0
        L[l].me1_b = F(pi++); L[l].me1_w = F(pi++);   // mlp_edges.l1
        L[l].mn0_b = F(pi++); L[l].mn0_w = F(pi++);   // mlp_nodes.l0
        L[l].mn1_b = F(pi++); L[l].mn1_w = F(pi++);   // mlp_nodes.l1
        L[l].no_b  = F(pi++); L[l].no_w  = F(pi++);   // nodes_out
        L[l].qkv_b = F(pi++); L[l].qkv_w = F(pi++);   // qkv
        L[l].ss_b  = F(pi++); L[l].ss_w  = F(pi++);   // ss
    }
    const float *lc_b = F(pi++),     *lc_w = F(pi++);
    const float *mic0_b = F(pi++), *mic0_w = F(pi++), *mic1_b = F(pi++), *mic1_w = F(pi++);
    const float *mie0_b = F(pi++), *mie0_w = F(pi++), *mie1_b = F(pi++), *mie1_w = F(pi++);
    const float *min0_b = F(pi++), *min0_w = F(pi++), *min1_b = F(pi++), *min1_w = F(pi++);
    const float *moe0_b = F(pi++), *moe0_w = F(pi++), *moe1_b = F(pi++), *moe1_w = F(pi++);
    const float *mop0_b = F(pi++), *mop0_w = F(pi++), *mop1_b = F(pi++), *mop1_w = F(pi++);
    const float *mot0_b = F(pi++), *mot0_w = F(pi++), *mot1_b = F(pi++), *mot1_w = F(pi++);
    const float *pos_embs  = F(pi++);
    const float *time_embs = F(pi++);

    // workspace layout (float offsets)
    float* Wf = (float*)d_ws;
    const long OFF_NODES = 0;            // 262144
    const long OFF_CONDS = 262144;       // 4096
    const long OFF_COND4 = 266240;       // 16384
    const long OFF_NNORM = 282624;       // 262144
    const long OFF_QKV   = 544768;       // 786432
    const long OFF_WV    = 1331200;      // 262144
    const long OFF_H1    = 1593344;      // 1048576 (also t1/p1)
    const long OFF_S     = 2641920;      // 524288
    const long OFF_AS    = 3166208;      // 524288
    const long OFF_TE    = 3690496;      // 4096
    const long OFF_C1    = 3694592;      // 4096
    const long OFF_COND2 = 3698688;      // 8192
    const long OFF_TYPES = 3706880;      // 6144
    const long OFF_PRMS  = 3713024;      // 14336
    const long OFF_EDGES = 3727360;      // 8388608
    const long OFF_ENORM = 12115968;     // 8388608
    const long OFF_E1    = 20504576;     // 8388608
    const long OFF_SS    = 28893184;     // 33554432
    const long OFF_NE    = 62447616;     // 16777216

    const int BN_ = MB * MN;            // 1024 node rows
    const int BNN = MB * MN * MN;       // 65536 edge rows

    // ---- conds path: te = time_embs[t]; conds = leaky(mlp2(te))
    tgather_kernel<<<MB, CHD, 0, stream>>>(timestep, time_embs, Wf + OFF_TE);
    launch_gemm(Wf + OFF_TE, mic0_w, mic0_b, nullptr, Wf + OFF_C1,    MB, CHD, CHD, 1, stream);
    launch_gemm(Wf + OFF_C1, mic1_w, mic1_b, nullptr, Wf + OFF_CONDS, MB, CHD, CHD, 1, stream);

    // ---- nodes embed: mlp2(21 -> 1024 -> 256) + pos_embs
    launch_gemm(nodes_in,    min0_w, min0_b, nullptr, Wf + OFF_H1,    BN_, 1024, 21,   1, stream);
    launch_gemm(Wf + OFF_H1, min1_w, min1_b, nullptr, Wf + OFF_NODES, BN_, NHD,  1024, 0, stream);
    posadd_kernel<<<(BN_ * NHD) / 256, 256, 0, stream>>>(Wf + OFF_NODES, pos_embs);

    // ---- edges embed: mlp2(5 -> 128 -> 128)
    launch_gemm(edges_in,    mie0_w, mie0_b, nullptr, Wf + OFF_E1,    BNN, EHD, 5,   1, stream);
    launch_gemm(Wf + OFF_E1, mie1_w, mie1_b, nullptr, Wf + OFF_EDGES, BNN, EHD, EHD, 0, stream);

    // ---- transformer layers
    for (int l = 0; l < 4; ++l) {
        // cond4 = lin_cond(conds): [16,1024] -> (m_in,a_in,m_out,a_out)
        launch_gemm(Wf + OFF_CONDS, L[l].lc_w, L[l].lc_b, nullptr, Wf + OFF_COND4, MB, 4 * NHD, CHD, 0, stream);
        // nodes_norm = leaky(ln(nodes)*(1+m_in)+a_in) ; edges_norm = leaky(ln(edges))
        ln_mod_kernel<<<BN_, 128, 0, stream>>>(Wf + OFF_NODES, Wf + OFF_NNORM,
                                               Wf + OFF_COND4, Wf + OFF_COND4 + NHD, NHD, MN, 4 * NHD);
        ln_mod_kernel<<<BNN, 128, 0, stream>>>(Wf + OFF_EDGES, Wf + OFF_ENORM,
                                               nullptr, nullptr, EHD, MN * MN, 0);
        // qkv and ss projections
        launch_gemm(Wf + OFF_NNORM, L[l].qkv_w, L[l].qkv_b, nullptr, Wf + OFF_QKV, BN_, 3 * NHD, NHD, 0, stream);
        launch_gemm(Wf + OFF_ENORM, L[l].ss_w,  L[l].ss_b,  nullptr, Wf + OFF_SS,  BNN, 2 * NHD, EHD, 0, stream);
        // attention elementwise + logits, softmax over i, A@V
        attn_kernel<<<dim3(MN * MN, MB), 256, 0, stream>>>(Wf + OFF_QKV, Wf + OFF_SS, Wf + OFF_NE, Wf + OFF_S);
        softmax_kernel<<<MB * NHEADS * MN, 64, 0, stream>>>(Wf + OFF_S, Wf + OFF_AS);
        wv_kernel<<<dim3(MN, MB), 256, 0, stream>>>(Wf + OFF_AS, Wf + OFF_QKV, Wf + OFF_WV);
        // nodes += nodes_out(wv) ; edges += edges_out(leaky(ne))
        launch_gemm(Wf + OFF_WV, L[l].no_w, L[l].no_b, Wf + OFF_NODES, Wf + OFF_NODES, BN_, NHD, NHD, 0, stream);
        launch_gemm(Wf + OFF_NE, L[l].eo_w, L[l].eo_b, Wf + OFF_EDGES, Wf + OFF_EDGES, BNN, EHD, NHD, 0, stream);
        // nodes MLP with out-modulation
        ln_mod_kernel<<<BN_, 128, 0, stream>>>(Wf + OFF_NODES, Wf + OFF_NNORM,
                                               Wf + OFF_COND4 + 2 * NHD, Wf + OFF_COND4 + 3 * NHD, NHD, MN, 4 * NHD);
        launch_gemm(Wf + OFF_NNORM, L[l].mn0_w, L[l].mn0_b, nullptr,         Wf + OFF_H1,    BN_, 4 * NHD, NHD,     1, stream);
        launch_gemm(Wf + OFF_H1,    L[l].mn1_w, L[l].mn1_b, Wf + OFF_NODES,  Wf + OFF_NODES, BN_, NHD,     4 * NHD, 0, stream);
        // edges MLP
        ln_mod_kernel<<<BNN, 128, 0, stream>>>(Wf + OFF_EDGES, Wf + OFF_ENORM, nullptr, nullptr, EHD, MN * MN, 0);
        launch_gemm(Wf + OFF_ENORM, L[l].me0_w, L[l].me0_b, nullptr,         Wf + OFF_E1,    BNN, EHD, EHD, 1, stream);
        launch_gemm(Wf + OFF_E1,    L[l].me1_w, L[l].me1_b, Wf + OFF_EDGES,  Wf + OFF_EDGES, BNN, EHD, EHD, 0, stream);
    }

    // ---- output heads
    // cond2 = lin_cond(conds): shift=[0:256], scale=[256:512]
    launch_gemm(Wf + OFF_CONDS, lc_w, lc_b, nullptr, Wf + OFF_COND2, MB, 2 * NHD, CHD, 0, stream);
    ln_mod_kernel<<<BN_, 128, 0, stream>>>(Wf + OFF_NODES, Wf + OFF_NNORM,
                                           Wf + OFF_COND2 + NHD, Wf + OFF_COND2, NHD, MN, 2 * NHD);
    // types head: 256 -> 1024 -> 6
    launch_gemm(Wf + OFF_NNORM, mot0_w, mot0_b, nullptr, Wf + OFF_H1,    BN_, 1024, NHD,  1, stream);
    launch_gemm(Wf + OFF_H1,    mot1_w, mot1_b, nullptr, Wf + OFF_TYPES, BN_, 6,    1024, 0, stream);
    // params head: 256 -> 1024 -> 14
    launch_gemm(Wf + OFF_NNORM, mop0_w, mop0_b, nullptr, Wf + OFF_H1,   BN_, 1024, NHD,  1, stream);
    launch_gemm(Wf + OFF_H1,    mop1_w, mop1_b, nullptr, Wf + OFF_PRMS, BN_, 14,   1024, 0, stream);
    assemble_nodes_kernel<<<BN_, 32, 0, stream>>>(Wf + OFF_TYPES, Wf + OFF_PRMS, (float*)d_out);
    // edges head: leaky(ln) -> 128 -> 128 -> 5, written straight into d_out
    ln_mod_kernel<<<BNN, 128, 0, stream>>>(Wf + OFF_EDGES, Wf + OFF_ENORM, nullptr, nullptr, EHD, MN * MN, 0);
    launch_gemm(Wf + OFF_ENORM, moe0_w, moe0_b, nullptr, Wf + OFF_E1, BNN, EHD, EHD, 1, stream);
    launch_gemm(Wf + OFF_E1, moe1_w, moe1_b, nullptr,
                (float*)d_out + (long)MB * MN * 21, BNN, 5, EHD, 0, stream);
}
